// UrbanGNN_13511967113428
// MI455X (gfx1250) — matrix operand order repaired and checked by reading, not compiled, not run on hardware
//
#include <hip/hip_runtime.h>

#define NN    50000
#define EE    800000
#define FIN   64
#define HDIM  128
#define CDIM  16
#define LAYERS 3

typedef __attribute__((ext_vector_type(16))) __bf16 v16bf;
typedef __attribute__((ext_vector_type(8)))  __bf16 v8bf;
typedef __attribute__((ext_vector_type(8)))  float  v8f;

// ---------------------------------------------------------------------------
// WMMA helpers (CDNA5, wave32).  D = A(16x32 bf16) * B(32x16 bf16) + C(16x16 f32)
// ---------------------------------------------------------------------------
__device__ __forceinline__ v8f wmma_bf16(v16bf a, v16bf b, v8f c) {
  // (neg_a, A, neg_b, B, c_mod, C, reuse_a, reuse_b)
  return __builtin_amdgcn_wmma_f32_16x16x32_bf16(false, a, false, b, (short)0, c,
                                                 false, false);
}

// Async global->LDS copy, 16B per lane (GLOBAL_LOAD_ASYNC_TO_LDS_B128, ASYNCcnt).
// lds_off is the workgroup-relative LDS byte offset (= low 32 bits of the
// generic pointer to a __shared__ object, per the flat-LDS aperture mapping).
__device__ __forceinline__ void async_copy_b128(unsigned lds_off, const void* gaddr) {
  asm volatile("global_load_async_to_lds_b128 %0, %1, off"
               :: "v"(lds_off), "v"(gaddr)
               : "memory");
}
__device__ __forceinline__ void wait_async0() {
  asm volatile("s_wait_asynccnt 0x0" ::: "memory");
}

// A fragment: 16x32 tile, row-major source with stride `ld` (bf16).
// ISA layout: lane L<16 -> row L, K = kb..kb+7 (v0-3) and kb+16..kb+23 (v4-7);
// lanes 16..31 -> same rows, kb += 8.
__device__ __forceinline__ v16bf load_a_frag(const __bf16* base, int ld, int k0, int lane) {
  int row = lane & 15;
  int kb  = k0 + ((lane >> 4) << 3);
  const __bf16* p = base + row * ld + kb;
  v8bf lo = *(const v8bf*)(p);        // K = kb .. kb+7
  v8bf hi = *(const v8bf*)(p + 16);   // K = kb+16 .. kb+23
  v16bf a;
#pragma unroll
  for (int j = 0; j < 8; ++j) { a[j] = lo[j]; a[8 + j] = hi[j]; }
  return a;
}

// B fragment: 32x16 tile of W^T where weights are stored [out][in] row-major
// (ld = in-dim).  ISA layout: lane L<16 -> col L, K = k0..k0+7 and k0+16..k0+23;
// lanes 16..31 -> same cols, K += 8.  Per-lane: 2x16B loads.
__device__ __forceinline__ v16bf load_b_frag(const __bf16* w, int ld, int n0, int k0, int lane) {
  int col  = n0 + (lane & 15);
  int koff = k0 + ((lane >> 4) << 3);
  const __bf16* p = w + col * ld + koff;
  v8bf lo = *(const v8bf*)(p);
  v8bf hi = *(const v8bf*)(p + 16);
  v16bf b;
#pragma unroll
  for (int j = 0; j < 8; ++j) { b[j] = lo[j]; b[8 + j] = hi[j]; }
  return b;
}

#define V8F_ZERO {0.f,0.f,0.f,0.f,0.f,0.f,0.f,0.f}

__device__ __forceinline__ float sigmoidf_fast(float x) {
  return 1.0f / (1.0f + __expf(-x));
}

// ---------------------------------------------------------------------------
// Kernel: convert weights to bf16.  W[l] is [H][H] (h @ W), store transposed
// as WT[l][n][k] so GEMM B-fragments are K-contiguous, like w_ih/w_hh already are.
// ---------------------------------------------------------------------------
__global__ void prep_weights(const float* __restrict__ W,
                             const float* __restrict__ wih,
                             const float* __restrict__ whh,
                             __bf16* __restrict__ WT,
                             __bf16* __restrict__ wihb,
                             __bf16* __restrict__ whhb) {
  int idx = blockIdx.x * blockDim.x + threadIdx.x;
  if (idx < LAYERS * HDIM * HDIM) {
    int l = idx / (HDIM * HDIM);
    int r = idx - l * HDIM * HDIM;
    int n = r >> 7, k = r & 127;
    WT[idx] = (__bf16)W[l * HDIM * HDIM + k * HDIM + n];
  }
  if (idx < 3 * HDIM * HDIM) {   // 384*128
    wihb[idx] = (__bf16)wih[idx];
    whhb[idx] = (__bf16)whh[idx];
  }
}

// Kernel: h = pad(x, H-F_IN) in fp32 + bf16 shadow copy.
__global__ void init_h(const float* __restrict__ x,
                       float* __restrict__ h, __bf16* __restrict__ hb) {
  int idx = blockIdx.x * blockDim.x + threadIdx.x;
  if (idx >= NN * HDIM) return;
  int n = idx >> 7, c = idx & 127;
  float v = (c < FIN) ? x[n * FIN + c] : 0.0f;
  h[idx]  = v;
  hb[idx] = (__bf16)v;
}

__global__ void zero_agg(float* __restrict__ agg) {
  int idx = blockIdx.x * blockDim.x + threadIdx.x;
  if (idx < NN * HDIM) agg[idx] = 0.0f;
}

// ---------------------------------------------------------------------------
// m = h @ W[l].  One block (8 waves) per 16-node tile; wave w -> cols [16w,16w+16).
// A tile staged once in LDS via async global->LDS copy, shared by all 8 waves.
// ---------------------------------------------------------------------------
__global__ __launch_bounds__(256) void msg_gemm(const __bf16* __restrict__ hb,
                                                const __bf16* __restrict__ WTl,
                                                float* __restrict__ m) {
  __shared__ alignas(16) __bf16 lds_h[16 * HDIM];
  int m0 = blockIdx.x << 4;
  int t  = threadIdx.x;

  // 16x128 bf16 tile: each of 256 threads asynchronously copies one 16B chunk.
  {
    unsigned lds_off = (unsigned)(size_t)(lds_h) + t * 16u;
    async_copy_b128(lds_off, hb + m0 * HDIM + t * 8);
    wait_async0();
  }
  __syncthreads();

  int lane = t & 31;
  int n0   = (t >> 5) << 4;            // wave id * 16
  v8f acc = V8F_ZERO;
#pragma unroll
  for (int k0 = 0; k0 < HDIM; k0 += 32) {
    v16bf a = load_a_frag(lds_h, HDIM, k0, lane);
    v16bf b = load_b_frag(WTl, HDIM, n0, k0, lane);
    acc = wmma_bf16(a, b, acc);
  }
  int col = lane & 15, rowb = (lane >> 4) << 3;
#pragma unroll
  for (int r = 0; r < 8; ++r)
    m[(m0 + rowb + r) * HDIM + n0 + col] = acc[r];
}

// ---------------------------------------------------------------------------
// agg[dst] += m[src].  One wave per edge: lane handles 4 channels (float4 read,
// 4x global_atomic_add_f32).  Coalesced 512B per wave.
// ---------------------------------------------------------------------------
__global__ void scatter_add(const float* __restrict__ m,
                            const int* __restrict__ ei,
                            float* __restrict__ agg) {
  int tid = blockIdx.x * blockDim.x + threadIdx.x;   // EE*32 = 25.6M < 2^31
  int e = tid >> 5;
  int c = (tid & 31) << 2;
  int s = ei[e];
  int d = ei[EE + e];
  float4 v = *(const float4*)(m + s * HDIM + c);
  float* ap = agg + d * HDIM + c;
  atomicAdd(ap + 0, v.x);
  atomicAdd(ap + 1, v.y);
  atomicAdd(ap + 2, v.z);
  atomicAdd(ap + 3, v.w);
}

// ---------------------------------------------------------------------------
// Fused GRUCell.  One block per 16-node tile.  Wave w owns column slice
// c0 = 16w of the H=128 hidden dim and computes the r/z/n slices of BOTH
// gi = agg@w_ihT and gh = h@w_hhT (6 accumulators, 24 WMMAs), then does the
// gate math and writes h (fp32) + bf16 shadow.  No gate matrices hit memory.
// h tile staged via async global->LDS copy; agg tile converted fp32->bf16
// through VGPRs (needs the conversion anyway).
// ---------------------------------------------------------------------------
__global__ __launch_bounds__(256) void gru_fused(const float* __restrict__ agg,
                                                 float* __restrict__ h,
                                                 __bf16* __restrict__ hb,
                                                 const __bf16* __restrict__ wihb,
                                                 const __bf16* __restrict__ whhb,
                                                 const float* __restrict__ bih,
                                                 const float* __restrict__ bhh) {
  __shared__ alignas(16) __bf16 lds_a[16 * HDIM];   // agg tile (bf16-converted)
  __shared__ alignas(16) __bf16 lds_h[16 * HDIM];   // h tile
  int m0 = blockIdx.x << 4;
  int t  = threadIdx.x;

  {
    // h tile: pure copy -> async global->LDS, 16B per thread.
    unsigned lds_off = (unsigned)(size_t)(lds_h) + t * 16u;
    async_copy_b128(lds_off, hb + m0 * HDIM + t * 8);

    // agg tile: load 8 floats, convert to bf16, store to LDS.
    const float4* af = (const float4*)(agg + m0 * HDIM);
    float4 f0 = af[2 * t], f1 = af[2 * t + 1];
    __bf16* la = lds_a + t * 8;
    la[0] = (__bf16)f0.x; la[1] = (__bf16)f0.y; la[2] = (__bf16)f0.z; la[3] = (__bf16)f0.w;
    la[4] = (__bf16)f1.x; la[5] = (__bf16)f1.y; la[6] = (__bf16)f1.z; la[7] = (__bf16)f1.w;

    wait_async0();
  }
  __syncthreads();

  int lane = t & 31;
  int c0   = (t >> 5) << 4;     // wave id * 16
  v8f ir = V8F_ZERO, iz = V8F_ZERO, in_ = V8F_ZERO;
  v8f hr = V8F_ZERO, hz = V8F_ZERO, hn  = V8F_ZERO;

#pragma unroll
  for (int k0 = 0; k0 < HDIM; k0 += 32) {
    v16bf aa = load_a_frag(lds_a, HDIM, k0, lane);
    v16bf ah = load_a_frag(lds_h, HDIM, k0, lane);
    ir  = wmma_bf16(aa, load_b_frag(wihb, HDIM, c0,        k0, lane), ir);
    iz  = wmma_bf16(aa, load_b_frag(wihb, HDIM, HDIM + c0, k0, lane), iz);
    in_ = wmma_bf16(aa, load_b_frag(wihb, HDIM, 2*HDIM+c0, k0, lane), in_);
    hr  = wmma_bf16(ah, load_b_frag(whhb, HDIM, c0,        k0, lane), hr);
    hz  = wmma_bf16(ah, load_b_frag(whhb, HDIM, HDIM + c0, k0, lane), hz);
    hn  = wmma_bf16(ah, load_b_frag(whhb, HDIM, 2*HDIM+c0, k0, lane), hn);
  }

  int col = lane & 15, rowb = (lane >> 4) << 3;
  int c = c0 + col;
  float bir = bih[c], biz = bih[HDIM + c], bin = bih[2 * HDIM + c];
  float bhr = bhh[c], bhz = bhh[HDIM + c], bhn = bhh[2 * HDIM + c];
#pragma unroll
  for (int r = 0; r < 8; ++r) {
    int row = m0 + rowb + r;
    float rg = sigmoidf_fast((ir[r] + bir) + (hr[r] + bhr));
    float zg = sigmoidf_fast((iz[r] + biz) + (hz[r] + bhz));
    float ng = tanhf((in_[r] + bin) + rg * (hn[r] + bhn));
    float ho = h[row * HDIM + c];
    float hv = (1.0f - zg) * ng + zg * ho;
    h[row * HDIM + c]  = hv;
    hb[row * HDIM + c] = (__bf16)hv;
  }
}

// ---------------------------------------------------------------------------
// out = log_softmax(h @ lin_wT + lin_b).  C=16 is tiny (0.6% of FLOPs):
// one thread per node, fp32 VALU.
// ---------------------------------------------------------------------------
__global__ void classifier(const float* __restrict__ h,
                           const float* __restrict__ lw,
                           const float* __restrict__ lb,
                           float* __restrict__ out) {
  int n = blockIdx.x * blockDim.x + threadIdx.x;
  if (n >= NN) return;
  const float* hrow = h + n * HDIM;
  float logits[CDIM];
#pragma unroll
  for (int c = 0; c < CDIM; ++c) {
    float s = lb[c];
    const float* w = lw + c * HDIM;
    for (int k = 0; k < HDIM; k += 4) {
      s += hrow[k] * w[k] + hrow[k+1] * w[k+1] + hrow[k+2] * w[k+2] + hrow[k+3] * w[k+3];
    }
    logits[c] = s;
  }
  float mx = logits[0];
#pragma unroll
  for (int c = 1; c < CDIM; ++c) mx = fmaxf(mx, logits[c]);
  float sum = 0.0f;
#pragma unroll
  for (int c = 0; c < CDIM; ++c) sum += __expf(logits[c] - mx);
  float lse = mx + __logf(sum);
#pragma unroll
  for (int c = 0; c < CDIM; ++c) out[n * CDIM + c] = logits[c] - lse;
}

// ---------------------------------------------------------------------------
extern "C" void kernel_launch(void* const* d_in, const int* in_sizes, int n_in,
                              void* d_out, int out_size, void* d_ws, size_t ws_size,
                              hipStream_t stream) {
  const float* x   = (const float*)d_in[0];
  const int*   ei  = (const int*)d_in[1];     // [2,E] int32 (JAX default int)
  const float* W   = (const float*)d_in[2];
  const float* wih = (const float*)d_in[3];
  const float* whh = (const float*)d_in[4];
  const float* bih = (const float*)d_in[5];
  const float* bhh = (const float*)d_in[6];
  const float* lw  = (const float*)d_in[7];
  const float* lb  = (const float*)d_in[8];
  float* out = (float*)d_out;

  char* ws = (char*)d_ws;
  size_t o = 0;
  auto take = [&](size_t bytes) {
    char* p = ws + o;
    o = (o + bytes + 255) & ~(size_t)255;
    return p;
  };
  float*  h    = (float*)take((size_t)NN * HDIM * 4);
  __bf16* hb   = (__bf16*)take((size_t)NN * HDIM * 2);
  float*  m    = (float*)take((size_t)NN * HDIM * 4);
  float*  agg  = (float*)take((size_t)NN * HDIM * 4);
  __bf16* WT   = (__bf16*)take((size_t)LAYERS * HDIM * HDIM * 2);
  __bf16* wihb = (__bf16*)take((size_t)3 * HDIM * HDIM * 2);
  __bf16* whhb = (__bf16*)take((size_t)3 * HDIM * HDIM * 2);

  prep_weights<<<(LAYERS * HDIM * HDIM + 255) / 256, 256, 0, stream>>>(
      W, wih, whh, WT, wihb, whhb);
  init_h<<<(NN * HDIM + 255) / 256, 256, 0, stream>>>(x, h, hb);

  for (int l = 0; l < LAYERS; ++l) {
    msg_gemm<<<NN / 16, 256, 0, stream>>>(hb, WT + l * HDIM * HDIM, m);
    zero_agg<<<(NN * HDIM + 255) / 256, 256, 0, stream>>>(agg);
    scatter_add<<<(EE * 32) / 256, 256, 0, stream>>>(m, ei, agg);
    gru_fused<<<NN / 16, 256, 0, stream>>>(agg, h, hb, wihb, whhb, bih, bhh);
  }
  classifier<<<(NN + 255) / 256, 256, 0, stream>>>(h, lw, lb, out);
}
